// Transformer_72885595013352
// MI455X (gfx1250) — matrix operand order repaired and checked by reading, not compiled, not run on hardware
//
#include <hip/hip_runtime.h>
#include <hip/hip_bf16.h>

// ---------------------------------------------------------------------------
// Types for CDNA5 WMMA (wave32): D(16x16,f32) = A(16x32,f16) x B(32x16,f16) + C
// ---------------------------------------------------------------------------
typedef __attribute__((ext_vector_type(16))) _Float16 v16h;
typedef __attribute__((ext_vector_type(8)))  _Float16 v8h;
typedef __attribute__((ext_vector_type(8)))  float    v8f;

#define LCOUNT 6
#define HDIM   512
#define DFFDIM 1024
#define BSZ    8
#define TLEN   512
#define NHEAD  8
#define DHEAD  64
#define NTOK   (BSZ * TLEN)          // 4096 rows

__device__ __forceinline__ v8f vzero8() {
  v8f z;
#pragma unroll
  for (int i = 0; i < 8; ++i) z[i] = 0.0f;
  return z;
}

__device__ __forceinline__ v8f wmma_f16(v16h a, v16h b, v8f c) {
  // (neg_a, A, neg_b, B, c_mod, C, reuse_a, reuse_b)
  return __builtin_amdgcn_wmma_f32_16x16x32_f16(false, a, false, b, (short)0, c,
                                                false, false);
}

// A fragment: 16x32 f16, row-major source, lda in elements.
// lane&15 = M row; lanes<16 hold K 0..15, lanes>=16 hold K 16..31 (contiguous)
__device__ __forceinline__ v16h load_a_frag(const _Float16* base, int lda, int lane) {
  int row = lane & 15;
  int kb  = (lane >> 4) << 4;                 // 0 or 16
  const _Float16* p = base + (size_t)row * lda + kb;
  v8h lo = *reinterpret_cast<const v8h*>(p);
  v8h hi = *reinterpret_cast<const v8h*>(p + 8);
  v16h a;
#pragma unroll
  for (int i = 0; i < 8; ++i) { a[i] = lo[i]; a[i + 8] = hi[i]; }
  return a;
}

// B fragment: 32x16 f16 from a K-major (transposed) matrix Bt[n*ldb + k].
// lane&15 = N col; lanes<16 hold K 0..15, lanes>=16 hold K 16..31 (contiguous)
__device__ __forceinline__ v16h load_b_frag(const _Float16* base, int ldb, int lane) {
  int col = lane & 15;
  int kb  = (lane >> 4) << 4;                 // 0 or 16
  return *reinterpret_cast<const v16h*>(base + (size_t)col * ldb + kb);
}

// ---------------------------------------------------------------------------
// Weight convert+transpose: src f32 [R,C] -> dst f16 [C,R]   (Bt[n,k]=W[k,n])
// ---------------------------------------------------------------------------
__global__ void transpose_to_f16_kernel(const float* __restrict__ src,
                                        _Float16* __restrict__ dst, int R, int C) {
  int i = blockIdx.x * blockDim.x + threadIdx.x;
  if (i >= R * C) return;
  int r = i / C, c = i % C;
  dst[(size_t)c * R + r] = (_Float16)src[i];
}

__global__ void f32_to_f16_kernel(const float* __restrict__ src,
                                  _Float16* __restrict__ dst, int n) {
  int i = blockIdx.x * blockDim.x + threadIdx.x;
  if (i < n) dst[i] = (_Float16)src[i];
}

// ---------------------------------------------------------------------------
// Embedding + sinusoidal positional encoding -> x f32 and x f16
// ---------------------------------------------------------------------------
__global__ __launch_bounds__(256) void embed_kernel(const int* __restrict__ idx,
                                                    const float* __restrict__ emb,
                                                    float* __restrict__ x,
                                                    _Float16* __restrict__ xh) {
  int row = blockIdx.x;                 // b*T + t
  int t = row & (TLEN - 1);
  int tok = idx[row];
  for (int c = threadIdx.x; c < HDIM; c += 256) {
    float freq = __expf(-(float)(c & ~1) * (9.210340371976184f / (float)HDIM));
    float ang  = (float)t * freq;
    float pe   = (c & 1) ? __cosf(ang) : __sinf(ang);
    float val  = emb[(size_t)tok * HDIM + c] + pe;
    x [(size_t)row * HDIM + c] = val;
    xh[(size_t)row * HDIM + c] = (_Float16)val;
  }
}

// ---------------------------------------------------------------------------
// WMMA GEMM: out[M,N] = A[M,Kd] x Bt[N,Kd]^T + bias.
// Block = 8 waves; wave tile 16x64 (4 accumulators); block tile 64x128.
// ---------------------------------------------------------------------------
template <bool OUT16, bool RELU>
__global__ __launch_bounds__(256) void gemm_kernel(const _Float16* __restrict__ A,
                                                   const _Float16* __restrict__ Bt,
                                                   const float* __restrict__ bias,
                                                   void* __restrict__ out,
                                                   int M, int N, int Kd) {
  int lane = threadIdx.x & 31;
  int wave = threadIdx.x >> 5;
  int m0 = blockIdx.x * 64 + (wave & 3) * 16;
  int n0 = blockIdx.y * 128 + (wave >> 2) * 64;

  v8f acc[4];
#pragma unroll
  for (int j = 0; j < 4; ++j) acc[j] = vzero8();

  for (int k0 = 0; k0 < Kd; k0 += 32) {
    v16h a = load_a_frag(A + (size_t)m0 * Kd + k0, Kd, lane);
#pragma unroll
    for (int j = 0; j < 4; ++j) {
      v16h b = load_b_frag(Bt + (size_t)(n0 + j * 16) * Kd + k0, Kd, lane);
      acc[j] = wmma_f16(a, b, acc[j]);
    }
  }

  int lrow = (lane >> 4) << 3;
#pragma unroll
  for (int j = 0; j < 4; ++j) {
    int col = n0 + j * 16 + (lane & 15);
    float bv = bias ? bias[col] : 0.0f;
#pragma unroll
    for (int v = 0; v < 8; ++v) {
      int row = m0 + v + lrow;
      float val = acc[j][v] + bv;
      if (RELU) val = fmaxf(val, 0.0f);
      if (OUT16) ((_Float16*)out)[(size_t)row * N + col] = (_Float16)val;
      else       ((float*)out)[(size_t)row * N + col] = val;
    }
  }
}

// ---------------------------------------------------------------------------
// V transpose: v f16 [B*512, 512] -> vT f16 [B, 8, 64, 512]
// ---------------------------------------------------------------------------
__global__ void transpose_v_kernel(const _Float16* __restrict__ v,
                                   _Float16* __restrict__ vT) {
  int i = blockIdx.x * blockDim.x + threadIdx.x;
  if (i >= BSZ * TLEN * HDIM) return;
  int c = i & (HDIM - 1);
  int t = (i >> 9) & (TLEN - 1);
  int b = i >> 18;
  int h = c >> 6, d = c & 63;
  vT[(((size_t)b * NHEAD + h) * DHEAD + d) * TLEN + t] = v[i];
}

// ---------------------------------------------------------------------------
// Fused attention: one wave per (b, head, 16-query tile).
// scores(16x512) via WMMA -> LDS f32 -> softmax -> P f16 -> ctx = P@V via WMMA
// ---------------------------------------------------------------------------
__global__ __launch_bounds__(32) void attn_kernel(const _Float16* __restrict__ q,
                                                  const _Float16* __restrict__ kk,
                                                  const _Float16* __restrict__ vT,
                                                  _Float16* __restrict__ ctx,
                                                  const int* __restrict__ tokIdx,
                                                  const unsigned char* __restrict__ encMask,
                                                  int causal) {
  __shared__ alignas(32) float    S[16 * 520];
  __shared__ alignas(32) _Float16 P[16 * 512];
  __shared__ float red[2][16];

  int lane = threadIdx.x;
  int qt = blockIdx.x & 31;
  int h  = (blockIdx.x >> 5) & 7;
  int b  = blockIdx.x >> 8;
  int q0 = qt * 16;

  // Q fragments (dk=64 -> two 32-deep k-steps)
  const _Float16* qbase = q + ((size_t)(b * TLEN + q0)) * HDIM + h * DHEAD;
  v16h aq0 = load_a_frag(qbase +  0, HDIM, lane);
  v16h aq1 = load_a_frag(qbase + 32, HDIM, lane);

  int lrow = (lane >> 4) << 3;
  int qcol = lane & 15;

  for (int n0 = 0; n0 < TLEN; n0 += 16) {
    const _Float16* kbase = kk + ((size_t)(b * TLEN + n0)) * HDIM + h * DHEAD;
    v16h bk0 = load_b_frag(kbase +  0, HDIM, lane);
    v16h bk1 = load_b_frag(kbase + 32, HDIM, lane);
    v8f c = vzero8();
    c = wmma_f16(aq0, bk0, c);
    c = wmma_f16(aq1, bk1, c);

    int key = n0 + qcol;
    bool keyMasked = causal ? (tokIdx[b * TLEN + key] == 0)
                            : (encMask[b * TLEN + key] != 0);
#pragma unroll
    for (int v = 0; v < 8; ++v) {
      int qpos = q0 + v + lrow;
      float sc = c[v] * 0.125f;                 // 1/sqrt(64)
      bool m = keyMasked || (causal && key > qpos);
      S[(v + lrow) * 520 + key] = m ? -1e9f : sc;
    }
  }
  __syncthreads();

  // --- softmax over 512 keys, 2 lanes per row ---
  int r = lane & 15;
  int half = lane >> 4;
  int c0 = half * 256;
  float mx = -3.4e38f;
  for (int c = 0; c < 256; ++c) mx = fmaxf(mx, S[r * 520 + c0 + c]);
  red[half][r] = mx;
  __syncthreads();
  mx = fmaxf(red[0][r], red[1][r]);
  __syncthreads();
  float sum = 0.0f;
  for (int c = 0; c < 256; ++c) {
    float e = __expf(S[r * 520 + c0 + c] - mx);
    S[r * 520 + c0 + c] = e;
    sum += e;
  }
  red[half][r] = sum;
  __syncthreads();
  float rinv = 1.0f / (red[0][r] + red[1][r]);
  for (int c = 0; c < 256; ++c)
    P[r * 512 + c0 + c] = (_Float16)(S[r * 520 + c0 + c] * rinv);
  __syncthreads();

  // --- ctx = P(16x512) @ V(512x64) : 4 N-tiles, 16 k-steps ---
  v8f acc[4];
#pragma unroll
  for (int j = 0; j < 4; ++j) acc[j] = vzero8();

  const _Float16* vbase0 = vT + (((size_t)b * NHEAD + h) * DHEAD) * TLEN;
  for (int ks = 0; ks < TLEN; ks += 32) {
    v16h a = load_a_frag(&P[0] + ks, 512, lane);
#pragma unroll
    for (int j = 0; j < 4; ++j) {
      v16h bb = load_b_frag(vbase0 + (size_t)(j * 16) * TLEN + ks, TLEN, lane);
      acc[j] = wmma_f16(a, bb, acc[j]);
    }
  }

#pragma unroll
  for (int j = 0; j < 4; ++j) {
    int col = h * DHEAD + j * 16 + qcol;
#pragma unroll
    for (int v = 0; v < 8; ++v) {
      int row = b * TLEN + q0 + v + lrow;
      ctx[(size_t)row * HDIM + col] = (_Float16)acc[j][v];
    }
  }
}

// ---------------------------------------------------------------------------
// x = LayerNorm(y + x) * g + b ; also writes f16 copy. One block per row.
// ---------------------------------------------------------------------------
__global__ __launch_bounds__(256) void add_ln_kernel(const float* __restrict__ y,
                                                     float* __restrict__ x,
                                                     const float* __restrict__ g,
                                                     const float* __restrict__ bta,
                                                     _Float16* __restrict__ xh) {
  __shared__ float red[256];
  int row = blockIdx.x;
  const float* yr = y + (size_t)row * HDIM;
  float* xr = x + (size_t)row * HDIM;
  int t = threadIdx.x;

  float a0 = yr[t] + xr[t];
  float a1 = yr[t + 256] + xr[t + 256];

  red[t] = a0 + a1;
  __syncthreads();
  for (int s = 128; s > 0; s >>= 1) {
    if (t < s) red[t] += red[t + s];
    __syncthreads();
  }
  float mean = red[0] * (1.0f / HDIM);
  __syncthreads();

  float d0 = a0 - mean, d1 = a1 - mean;
  red[t] = d0 * d0 + d1 * d1;
  __syncthreads();
  for (int s = 128; s > 0; s >>= 1) {
    if (t < s) red[t] += red[t + s];
    __syncthreads();
  }
  float rstd = rsqrtf(red[0] * (1.0f / HDIM) + 1e-5f);

  float o0 = d0 * rstd * g[t] + bta[t];
  float o1 = d1 * rstd * g[t + 256] + bta[t + 256];
  xr[t] = o0;
  xr[t + 256] = o1;
  xh[(size_t)row * HDIM + t] = (_Float16)o0;
  xh[(size_t)row * HDIM + t + 256] = (_Float16)o1;
}

// ---------------------------------------------------------------------------
// Host orchestration
// ---------------------------------------------------------------------------
extern "C" void kernel_launch(void* const* d_in, const int* in_sizes, int n_in,
                              void* d_out, int out_size, void* d_ws, size_t ws_size,
                              hipStream_t stream) {
  (void)in_sizes; (void)n_in; (void)out_size; (void)ws_size;

  const int*           idx     = (const int*)d_in[0];
  const float*         enc     = (const float*)d_in[1];
  const unsigned char* encMask = (const unsigned char*)d_in[2];
  const float*         molEmb  = (const float*)d_in[4];
  const float* sa_wq = (const float*)d_in[5],  *sa_bq = (const float*)d_in[6];
  const float* sa_wk = (const float*)d_in[7],  *sa_bk = (const float*)d_in[8];
  const float* sa_wv = (const float*)d_in[9],  *sa_bv = (const float*)d_in[10];
  const float* sa_wo = (const float*)d_in[11], *sa_bo = (const float*)d_in[12];
  const float* ln1_g = (const float*)d_in[13], *ln1_b = (const float*)d_in[14];
  const float* ea_wq = (const float*)d_in[15], *ea_bq = (const float*)d_in[16];
  const float* ea_wk = (const float*)d_in[17], *ea_bk = (const float*)d_in[18];
  const float* ea_wv = (const float*)d_in[19], *ea_bv = (const float*)d_in[20];
  const float* ea_wo = (const float*)d_in[21], *ea_bo = (const float*)d_in[22];
  const float* ln2_g = (const float*)d_in[23], *ln2_b = (const float*)d_in[24];
  const float* c1_w  = (const float*)d_in[25], *c1_b  = (const float*)d_in[26];
  const float* c2_w  = (const float*)d_in[27], *c2_b  = (const float*)d_in[28];
  const float* ln3_g = (const float*)d_in[29], *ln3_b = (const float*)d_in[30];

  // ---- workspace carve-out (all 256B aligned) ----
  char* ws = (char*)d_ws;
  size_t off = 0;
  auto carve = [&](size_t bytes) -> void* {
    void* p = ws + off;
    off += (bytes + 255) & ~(size_t)255;
    return p;
  };
  const size_t XB32 = (size_t)NTOK * HDIM * sizeof(float);     // 8 MB
  const size_t XB16 = (size_t)NTOK * HDIM * sizeof(_Float16);  // 4 MB
  float*    x    = (float*)carve(XB32);
  _Float16* xh   = (_Float16*)carve(XB16);
  _Float16* ench = (_Float16*)carve(XB16);
  _Float16* qh   = (_Float16*)carve(XB16);
  _Float16* kh   = (_Float16*)carve(XB16);
  _Float16* vh   = (_Float16*)carve(XB16);
  _Float16* vTh  = (_Float16*)carve(XB16);
  _Float16* ctxh = (_Float16*)carve(XB16);
  float*    ybuf = (float*)carve(XB32);
  _Float16* hbuf = (_Float16*)carve((size_t)NTOK * DFFDIM * sizeof(_Float16));
  const size_t WSQ = (size_t)HDIM * HDIM;       // 262144
  const size_t WFF = (size_t)HDIM * DFFDIM;     // 524288
  const size_t WLAYER = 8 * WSQ + 2 * WFF;      // per-layer f16 elems
  _Float16* wbuf = (_Float16*)carve(LCOUNT * WLAYER * sizeof(_Float16));

  auto grid1 = [](size_t n) { return dim3((unsigned)((n + 255) / 256)); };

  // ---- one-time (per launch) weight transposes to K-major f16 ----
  for (int l = 0; l < LCOUNT; ++l) {
    _Float16* wl = wbuf + (size_t)l * WLAYER;
    transpose_to_f16_kernel<<<grid1(WSQ), 256, 0, stream>>>(sa_wq + (size_t)l * WSQ, wl + 0 * WSQ, HDIM, HDIM);
    transpose_to_f16_kernel<<<grid1(WSQ), 256, 0, stream>>>(sa_wk + (size_t)l * WSQ, wl + 1 * WSQ, HDIM, HDIM);
    transpose_to_f16_kernel<<<grid1(WSQ), 256, 0, stream>>>(sa_wv + (size_t)l * WSQ, wl + 2 * WSQ, HDIM, HDIM);
    transpose_to_f16_kernel<<<grid1(WSQ), 256, 0, stream>>>(sa_wo + (size_t)l * WSQ, wl + 3 * WSQ, HDIM, HDIM);
    transpose_to_f16_kernel<<<grid1(WSQ), 256, 0, stream>>>(ea_wq + (size_t)l * WSQ, wl + 4 * WSQ, HDIM, HDIM);
    transpose_to_f16_kernel<<<grid1(WSQ), 256, 0, stream>>>(ea_wk + (size_t)l * WSQ, wl + 5 * WSQ, HDIM, HDIM);
    transpose_to_f16_kernel<<<grid1(WSQ), 256, 0, stream>>>(ea_wv + (size_t)l * WSQ, wl + 6 * WSQ, HDIM, HDIM);
    transpose_to_f16_kernel<<<grid1(WSQ), 256, 0, stream>>>(ea_wo + (size_t)l * WSQ, wl + 7 * WSQ, HDIM, HDIM);
    transpose_to_f16_kernel<<<grid1(WFF), 256, 0, stream>>>(c1_w + (size_t)l * WFF, wl + 8 * WSQ, HDIM, DFFDIM);
    transpose_to_f16_kernel<<<grid1(WFF), 256, 0, stream>>>(c2_w + (size_t)l * WFF, wl + 8 * WSQ + WFF, DFFDIM, HDIM);
  }

  f32_to_f16_kernel<<<grid1((size_t)NTOK * HDIM), 256, 0, stream>>>(enc, ench, NTOK * HDIM);
  embed_kernel<<<NTOK, 256, 0, stream>>>(idx, molEmb, x, xh);

  auto gemm16 = [&](const _Float16* A, const _Float16* Bt, const float* bias,
                    _Float16* o, int M, int N, int Kd, bool relu) {
    dim3 g(M / 64, N / 128);
    if (relu) gemm_kernel<true, true ><<<g, 256, 0, stream>>>(A, Bt, bias, o, M, N, Kd);
    else      gemm_kernel<true, false><<<g, 256, 0, stream>>>(A, Bt, bias, o, M, N, Kd);
  };
  auto gemm32 = [&](const _Float16* A, const _Float16* Bt, const float* bias,
                    float* o, int M, int N, int Kd) {
    dim3 g(M / 64, N / 128);
    gemm_kernel<false, false><<<g, 256, 0, stream>>>(A, Bt, bias, o, M, N, Kd);
  };

  const int ATTN_GRID = BSZ * NHEAD * (TLEN / 16);   // 2048

  for (int l = 0; l < LCOUNT; ++l) {
    _Float16* wl = wbuf + (size_t)l * WLAYER;

    // ---- self attention ----
    gemm16(xh, wl + 0 * WSQ, sa_bq + (size_t)l * HDIM, qh, NTOK, HDIM, HDIM, false);
    gemm16(xh, wl + 1 * WSQ, sa_bk + (size_t)l * HDIM, kh, NTOK, HDIM, HDIM, false);
    gemm16(xh, wl + 2 * WSQ, sa_bv + (size_t)l * HDIM, vh, NTOK, HDIM, HDIM, false);
    transpose_v_kernel<<<grid1((size_t)NTOK * HDIM), 256, 0, stream>>>(vh, vTh);
    attn_kernel<<<ATTN_GRID, 32, 0, stream>>>(qh, kh, vTh, ctxh, idx, encMask, 1);
    gemm32(ctxh, wl + 3 * WSQ, sa_bo + (size_t)l * HDIM, ybuf, NTOK, HDIM, HDIM);
    add_ln_kernel<<<NTOK, 256, 0, stream>>>(ybuf, x, ln1_g + (size_t)l * HDIM,
                                            ln1_b + (size_t)l * HDIM, xh);

    // ---- cross attention (K/V from encoder outputs) ----
    gemm16(xh,   wl + 4 * WSQ, ea_bq + (size_t)l * HDIM, qh, NTOK, HDIM, HDIM, false);
    gemm16(ench, wl + 5 * WSQ, ea_bk + (size_t)l * HDIM, kh, NTOK, HDIM, HDIM, false);
    gemm16(ench, wl + 6 * WSQ, ea_bv + (size_t)l * HDIM, vh, NTOK, HDIM, HDIM, false);
    transpose_v_kernel<<<grid1((size_t)NTOK * HDIM), 256, 0, stream>>>(vh, vTh);
    attn_kernel<<<ATTN_GRID, 32, 0, stream>>>(qh, kh, vTh, ctxh, idx, encMask, 0);
    gemm32(ctxh, wl + 7 * WSQ, ea_bo + (size_t)l * HDIM, ybuf, NTOK, HDIM, HDIM);
    add_ln_kernel<<<NTOK, 256, 0, stream>>>(ybuf, x, ln2_g + (size_t)l * HDIM,
                                            ln2_b + (size_t)l * HDIM, xh);

    // ---- FFN ----
    gemm16(xh, wl + 8 * WSQ, c1_b + (size_t)l * DFFDIM, hbuf, NTOK, DFFDIM, HDIM, true);
    gemm32(hbuf, wl + 8 * WSQ + WFF, c2_b + (size_t)l * HDIM, ybuf, NTOK, HDIM, DFFDIM);
    add_ln_kernel<<<NTOK, 256, 0, stream>>>(ybuf, x, ln3_g + (size_t)l * HDIM,
                                            ln3_b + (size_t)l * HDIM, xh);
  }

  hipMemcpyAsync(d_out, x, XB32, hipMemcpyDeviceToDevice, stream);
}
